// RNNModel_27797028340315
// MI455X (gfx1250) — matrix operand order repaired and checked by reading, not compiled, not run on hardware
//
// CDNA5 / gfx1250 bidirectional attention-LSTM.
// bf16 weights pre-swizzled into WMMA-fragment-major layout (1KB per 16x16x32
// B-fragment => 2x global_load_b128 per lane), V_WMMA_F32_16X16X32_BF16 for all
// per-step GEMMs, weights L2-resident; gate GEMM fused 4-wide (shared A, 4
// independent WMMA accumulator chains), attention GEMM fused 5-wide (shared B);
// persistent 8-WG-per-direction phase pipeline with device-scope arrive
// barriers; linear-pooling identity so no [B,T,H] outputs are materialized.
#include <hip/hip_runtime.h>
#include <hip/hip_bf16.h>

typedef __bf16 bf16;
typedef __attribute__((ext_vector_type(16))) __bf16 v16bf;
typedef __attribute__((ext_vector_type(8)))  float  v8f;

union Frag16 { v16bf bf; uint4 u[2]; };

constexpr int B = 32, T = 512, F = 256, H = 1024, C = 60, L = 5;
constexpr int FH = F + H;              // 1280
constexpr int NWG = 8;                 // workgroups per direction
constexpr int THREADS = 512;           // 16 wave32 per WG
constexpr int NT = NWG * THREADS;      // 4096 threads per direction
constexpr int NWAVE = NT / 32;         // 128 waves per direction

// ---------------- workspace layout (all 256B-aligned sizes) ----------------
constexpr size_t OFF_BAR  = 0;
constexpr size_t SZ_BAR   = 256;
constexpr size_t SZ_BH_F  = (size_t)2 * B * H * 4;
constexpr size_t SZ_BH_H  = (size_t)2 * B * H * 2;
constexpr size_t OFF_CST  = OFF_BAR + SZ_BAR;                    // c state  f32
constexpr size_t OFF_S    = OFF_CST + SZ_BH_F;                   // out sums f32
constexpr size_t OFF_HBF  = OFF_S   + SZ_BH_F;                   // h  bf16
constexpr size_t OFF_ATB  = OFF_HBF + SZ_BH_H;                   // attns bf16
constexpr size_t SZ_AST   = (size_t)2 * B * L * H * 2;
constexpr size_t OFF_AST  = OFF_ATB + SZ_BH_H;                   // attn_states bf16
constexpr size_t OFF_ZEND = OFF_AST + SZ_AST;                    // zeroed range end
constexpr size_t OFF_Q    = OFF_ZEND;                            // q    f32
constexpr size_t SZ_HF    = (size_t)2 * B * L * H * 4;
constexpr size_t OFF_HF   = OFF_Q   + SZ_BH_F;                   // hf   f32
constexpr size_t OFF_OUT  = OFF_HF  + SZ_HF;                     // out  f32
constexpr size_t OFF_SB   = OFF_OUT + SZ_BH_F;                   // scores f32
constexpr size_t SZ_SB    = 2048;
constexpr size_t OFF_D    = OFF_SB  + SZ_SB;                     // d (new attns) f32
constexpr size_t SZ_C1280 = (size_t)2 * B * FH * 2;
constexpr size_t SZ_C2048 = (size_t)2 * B * 2 * H * 2;
constexpr size_t OFF_CIN  = OFF_D   + SZ_BH_F;                   // [x|attns] bf16
constexpr size_t OFF_CZ   = OFF_CIN + SZ_C1280;                  // [xa|h]    bf16
constexpr size_t OFF_CQ   = OFF_CZ  + SZ_C1280;                  // [nc|nh]   bf16
constexpr size_t OFF_CP   = OFF_CQ  + SZ_C2048;                  // [nh|d]    bf16
constexpr size_t SZ_XBF   = (size_t)2 * T * B * F * 2;
constexpr size_t OFF_XBF  = OFF_CP  + SZ_C2048;                  // time-major x bf16
constexpr size_t OFF_W    = OFF_XBF + SZ_XBF;                    // packed bf16 weights
constexpr size_t EW_I = 0;
constexpr size_t EW_L = EW_I + (size_t)FH * F;
constexpr size_t EW_K = EW_L + (size_t)FH * 4 * H;
constexpr size_t EW_Q = EW_K + (size_t)H * H;
constexpr size_t EW_P = EW_Q + (size_t)2 * H * H;
constexpr size_t EW_DIR = EW_P + (size_t)2 * H * H;

// ------------------------------- helpers -----------------------------------
__device__ inline float sigm(float x) { return 1.0f / (1.0f + __expf(-x)); }

#define WMMA_BF16(A_, B_, C_)                                                  \
  __builtin_amdgcn_wmma_f32_16x16x32_bf16(false, (A_), false, (B_), (short)0,  \
                                          (C_), false, false)

// Single-tile GEMM: A bf16 row-major (16B-aligned rows), Wpk fragment-major.
__device__ inline v8f wmma_gemm(const bf16* __restrict__ A, int lda, int m0,
                                const bf16* __restrict__ Wpk, int N, int nt,
                                int K) {
  v8f acc = {0.f, 0.f, 0.f, 0.f, 0.f, 0.f, 0.f, 0.f};
  const int lane = threadIdx.x & 31;
  const int half = (lane < 16) ? 0 : 8;
  const bf16* arow  = A + (size_t)(m0 + (lane & 15)) * lda;
  const bf16* bptr  = Wpk + (size_t)nt * 512 + lane * 16;
  const size_t bstr = (size_t)(N / 16) * 512;
  for (int k0 = 0; k0 < K; k0 += 32) {
    __builtin_prefetch(bptr + bstr, 0, 1);
    Frag16 a, b;
    a.u[0] = *(const uint4*)(arow + k0 + half);
    a.u[1] = *(const uint4*)(arow + k0 + 16 + half);
    b.u[0] = *(const uint4*)(bptr);
    b.u[1] = *(const uint4*)(bptr + 8);
    acc = WMMA_BF16(a.bf, b.bf, acc);
    bptr += bstr;
  }
  return acc;
}

// 4-wide N-batched GEMM (shared A fragment, 4 packed-B streams, 4 independent
// accumulator chains) for the i/j/f/o gate tiles.
__device__ inline void wmma_gate4(const bf16* __restrict__ A, int lda, int m0,
                                  const bf16* __restrict__ Wpk, int N, int nt0,
                                  int ntstep, int K, v8f& r0, v8f& r1, v8f& r2,
                                  v8f& r3) {
  const int lane = threadIdx.x & 31;
  const int half = (lane < 16) ? 0 : 8;
  const bf16* arow  = A + (size_t)(m0 + (lane & 15)) * lda;
  const size_t bstr = (size_t)(N / 16) * 512;
  const size_t boff = (size_t)ntstep * 512;
  const bf16* bp    = Wpk + (size_t)nt0 * 512 + lane * 16;
  for (int k0 = 0; k0 < K; k0 += 32) {
    Frag16 a, b0, b1, b2, b3;
    a.u[0] = *(const uint4*)(arow + k0 + half);
    a.u[1] = *(const uint4*)(arow + k0 + 16 + half);
    __builtin_prefetch(bp + bstr, 0, 1);
    __builtin_prefetch(bp + boff + bstr, 0, 1);
    __builtin_prefetch(bp + 2 * boff + bstr, 0, 1);
    __builtin_prefetch(bp + 3 * boff + bstr, 0, 1);
    b0.u[0] = *(const uint4*)(bp);
    b0.u[1] = *(const uint4*)(bp + 8);
    b1.u[0] = *(const uint4*)(bp + boff);
    b1.u[1] = *(const uint4*)(bp + boff + 8);
    b2.u[0] = *(const uint4*)(bp + 2 * boff);
    b2.u[1] = *(const uint4*)(bp + 2 * boff + 8);
    b3.u[0] = *(const uint4*)(bp + 3 * boff);
    b3.u[1] = *(const uint4*)(bp + 3 * boff + 8);
    r0 = WMMA_BF16(a.bf, b0.bf, r0);
    r1 = WMMA_BF16(a.bf, b1.bf, r1);
    r2 = WMMA_BF16(a.bf, b2.bf, r2);
    r3 = WMMA_BF16(a.bf, b3.bf, r3);
    bp += bstr;
  }
}

// 5-wide M-batched GEMM (shared B fragment, 5 A row-streams, 5 independent
// accumulator chains) for the attention-window projection hf = astates @ Wk.
__device__ inline void wmma_hf5(const bf16* __restrict__ A, int lda, int m0,
                                int mstep, const bf16* __restrict__ Wpk, int N,
                                int nt, int K, v8f& r0, v8f& r1, v8f& r2,
                                v8f& r3, v8f& r4) {
  const int lane = threadIdx.x & 31;
  const int half = (lane < 16) ? 0 : 8;
  const bf16* arow   = A + (size_t)(m0 + (lane & 15)) * lda;
  const size_t astep = (size_t)mstep * lda;
  const size_t bstr  = (size_t)(N / 16) * 512;
  const bf16* bp     = Wpk + (size_t)nt * 512 + lane * 16;
  for (int k0 = 0; k0 < K; k0 += 32) {
    Frag16 b;
    __builtin_prefetch(bp + bstr, 0, 1);
    b.u[0] = *(const uint4*)(bp);
    b.u[1] = *(const uint4*)(bp + 8);
    const bf16* ar = arow + k0 + half;
    Frag16 a0, a1, a2, a3, a4;
    a0.u[0] = *(const uint4*)(ar);
    a0.u[1] = *(const uint4*)(ar + 16);
    a1.u[0] = *(const uint4*)(ar + astep);
    a1.u[1] = *(const uint4*)(ar + astep + 16);
    a2.u[0] = *(const uint4*)(ar + 2 * astep);
    a2.u[1] = *(const uint4*)(ar + 2 * astep + 16);
    a3.u[0] = *(const uint4*)(ar + 3 * astep);
    a3.u[1] = *(const uint4*)(ar + 3 * astep + 16);
    a4.u[0] = *(const uint4*)(ar + 4 * astep);
    a4.u[1] = *(const uint4*)(ar + 4 * astep + 16);
    r0 = WMMA_BF16(a0.bf, b.bf, r0);
    r1 = WMMA_BF16(a1.bf, b.bf, r1);
    r2 = WMMA_BF16(a2.bf, b.bf, r2);
    r3 = WMMA_BF16(a3.bf, b.bf, r3);
    r4 = WMMA_BF16(a4.bf, b.bf, r4);
    bp += bstr;
  }
}

__device__ inline void dir_barrier(unsigned* bar, unsigned& gen) {
  __syncthreads();
  if (threadIdx.x == 0) {
    __threadfence();
    unsigned target = (++gen) * (unsigned)NWG;
    atomicAdd(bar, 1u);
    while (__atomic_load_n(bar, __ATOMIC_ACQUIRE) < target)
      __builtin_amdgcn_s_sleep(1);
    __threadfence();
  }
  __syncthreads();
}

// ------------------------------- prep kernels ------------------------------
__global__ void zero_ws_kernel(char* ws) {
  unsigned* p = (unsigned*)ws;
  size_t n = OFF_ZEND / 4;
  for (size_t i = blockIdx.x * (size_t)blockDim.x + threadIdx.x; i < n;
       i += (size_t)gridDim.x * blockDim.x)
    p[i] = 0u;
}

// fp32 row-major [K][N] -> bf16 fragment-major packed layout:
// dst[((kt*(N/16)+nt)*32 + lane)*16 + j] =
//   src[(kt*32 + (j&8)*2 + (lane<16?0:8) + (j&7))*N + nt*16 + (lane&15)]
__global__ void pack_weight_kernel(const float* __restrict__ src,
                                   bf16* __restrict__ dst, int K, int N) {
  size_t n = (size_t)K * N;
  int ntiles = N / 16;
  for (size_t i = blockIdx.x * (size_t)blockDim.x + threadIdx.x; i < n;
       i += (size_t)gridDim.x * blockDim.x) {
    int j    = (int)(i & 15);
    int lane = (int)((i >> 4) & 31);
    size_t frag = i >> 9;
    int nt = (int)(frag % ntiles);
    int kt = (int)(frag / ntiles);
    int k  = kt * 32 + (j & 8) * 2 + ((lane < 16) ? 0 : 8) + (j & 7);
    int nn = nt * 16 + (lane & 15);
    dst[i] = (bf16)src[(size_t)k * N + nn];
  }
}

// time-major bf16 x, backward direction pre-reversed by per-batch seq_len
__global__ void build_x_kernel(const float* __restrict__ x,
                               const int* __restrict__ length,
                               bf16* __restrict__ xbf) {
  size_t n = (size_t)2 * T * B * F;
  for (size_t i = blockIdx.x * (size_t)blockDim.x + threadIdx.x; i < n;
       i += (size_t)gridDim.x * blockDim.x) {
    int f = (int)(i % F);
    int b = (int)((i / F) % B);
    int t = (int)((i / ((size_t)F * B)) % T);
    int d = (int)(i / ((size_t)F * B * T));
    int sl = length[b] / 4 + 1;
    int ts = (d == 0) ? t : ((t < sl) ? (sl - 1 - t) : t);
    xbf[i] = (bf16)x[((size_t)b * T + ts) * F + f];
  }
}

// ------------------------- persistent recurrence ---------------------------
__global__ __launch_bounds__(THREADS) void recurrence_kernel(
    char* ws, const int* __restrict__ length,
    const float* bi_f, const float* bl_f, const float* bq_f, const float* v_f,
    const float* bp_f, const float* bi_b, const float* bl_b, const float* bq_b,
    const float* v_b, const float* bp_b) {
  const int dir  = blockIdx.x / NWG;
  const int wg   = blockIdx.x % NWG;
  const int tidd = wg * THREADS + threadIdx.x;   // 0..NT-1 within direction
  const int gw   = tidd >> 5;                    // 0..127 wave id in direction
  const int lane = threadIdx.x & 31;

  unsigned* bar = (unsigned*)(ws + OFF_BAR) + dir;
  unsigned gen = 0;

  float* cst  = (float*)(ws + OFF_CST) + (size_t)dir * B * H;
  float* Ssum = (float*)(ws + OFF_S)   + (size_t)dir * B * H;
  float* qbuf = (float*)(ws + OFF_Q)   + (size_t)dir * B * H;
  float* hfb  = (float*)(ws + OFF_HF)  + (size_t)dir * B * L * H;
  float* outb = (float*)(ws + OFF_OUT) + (size_t)dir * B * H;
  float* sbuf = (float*)(ws + OFF_SB)  + (size_t)dir * B * L;
  float* dbuf = (float*)(ws + OFF_D)   + (size_t)dir * B * H;
  bf16* hbf   = (bf16*)(ws + OFF_HBF)  + (size_t)dir * B * H;
  bf16* atbf  = (bf16*)(ws + OFF_ATB)  + (size_t)dir * B * H;
  bf16* ast   = (bf16*)(ws + OFF_AST)  + (size_t)dir * B * L * H;   // [B*L][H]
  bf16* cIN   = (bf16*)(ws + OFF_CIN)  + (size_t)dir * B * FH;
  bf16* cZ    = (bf16*)(ws + OFF_CZ)   + (size_t)dir * B * FH;
  bf16* cQ    = (bf16*)(ws + OFF_CQ)   + (size_t)dir * B * 2 * H;
  bf16* cP    = (bf16*)(ws + OFF_CP)   + (size_t)dir * B * 2 * H;
  const bf16* xb = (bf16*)(ws + OFF_XBF) + (size_t)dir * T * B * F;
  const bf16* WB = (bf16*)(ws + OFF_W) + (size_t)dir * EW_DIR;
  const bf16 *Wi = WB + EW_I, *Wl = WB + EW_L, *Wk = WB + EW_K,
             *Wq = WB + EW_Q, *Wp = WB + EW_P;
  const float* bi = dir ? bi_b : bi_f;
  const float* bl = dir ? bl_b : bl_f;
  const float* bq = dir ? bq_b : bq_f;
  const float* vv = dir ? v_b  : v_f;
  const float* bp = dir ? bp_b : bp_f;

  for (int t = 0; t < T; ++t) {
    const bf16* xt = xb + (size_t)t * B * F;

    // ---- P0: cIN = [x_t | attns], cZ[:,F:] = h_{t-1} -----------------------
    for (int i = tidd; i < B * FH; i += NT) {
      int b = i / FH, k = i % FH;
      cIN[i] = (k < F) ? xt[b * F + k] : atbf[b * H + (k - F)];
    }
    for (int i = tidd; i < B * H; i += NT) {
      int b = i / H, k = i % H;
      cZ[b * FH + F + k] = hbf[i];
    }
    dir_barrier(bar, gen);

    // ---- P1: xa = cIN @ Wi + bi  -> bf16 into cZ[:, :F]  (2x16 tiles) ------
    if (gw < 32) {
      int mt = gw >> 4, nt = gw & 15;
      v8f acc = wmma_gemm(cIN, FH, mt * 16, Wi, F, nt, FH);
      int n  = nt * 16 + (lane & 15);
      int rb = mt * 16 + ((lane < 16) ? 0 : 8);
      float bn = bi[n];
#pragma unroll
      for (int v = 0; v < 8; ++v)
        cZ[(rb + v) * FH + n] = (bf16)(acc[v] + bn);
    }
    dir_barrier(bar, gen);

    // ---- P2: gates (128 tile-groups): zg = cZ @ Wl + bl, 4-wide fused ------
    {
      int mt = gw >> 6, ht = gw & 63;
      v8f ai = {0.f, 0.f, 0.f, 0.f, 0.f, 0.f, 0.f, 0.f};
      v8f aj = ai, af = ai, ao = ai;
      wmma_gate4(cZ, FH, mt * 16, Wl, 4 * H, ht, 64, FH, ai, aj, af, ao);
      int n  = ht * 16 + (lane & 15);
      int rb = mt * 16 + ((lane < 16) ? 0 : 8);
      float bli = bl[n], blj = bl[H + n], blf = bl[2 * H + n], blo = bl[3 * H + n];
#pragma unroll
      for (int v = 0; v < 8; ++v) {
        int b = rb + v;
        bool m = t < (length[b] / 4 + 1);
        float c_old = cst[b * H + n];
        float iv = sigm(ai[v] + bli);
        float jv = tanhf(aj[v] + blj);
        float fv = sigm(af[v] + blf + 1.0f);   // forget_bias
        float ov = sigm(ao[v] + blo);
        float nc = c_old * fv + iv * jv;
        float nh = tanhf(nc) * ov;
        bf16 hold = hbf[b * H + n];
        cst[b * H + n] = m ? nc : c_old;                 // frozen carry
        hbf[b * H + n] = m ? (bf16)nh : hold;
        cQ[b * 2 * H + n]     = (bf16)nc;                // raw for attention
        cQ[b * 2 * H + H + n] = (bf16)nh;
        cP[b * 2 * H + n]     = (bf16)nh;                // raw for out-proj
      }
    }
    dir_barrier(bar, gen);

    // ---- P3: q = cQ @ Wq + bq (128 tiles); hf = astates @ Wk, 5-wide -------
    {
      int mt = gw >> 6, nt = gw & 63;
      v8f acc = wmma_gemm(cQ, 2 * H, mt * 16, Wq, H, nt, 2 * H);
      int n  = nt * 16 + (lane & 15);
      int rb = mt * 16 + ((lane < 16) ? 0 : 8);
      float bn = bq[n];
#pragma unroll
      for (int v = 0; v < 8; ++v)
        qbuf[(rb + v) * H + n] = acc[v] + bn;
    }
    {
      // 640 tiles as 128 x (5 M-tiles sharing one B stream):
      // wave gw covers (mt, nt) for mt = mt0, mt0+2, ..., mt0+8 ; nt = gw%64
      int mt0 = gw >> 6, nt = gw & 63;
      v8f h0 = {0.f, 0.f, 0.f, 0.f, 0.f, 0.f, 0.f, 0.f};
      v8f h1 = h0, h2 = h0, h3 = h0, h4 = h0;
      wmma_hf5(ast, H, mt0 * 16, 32, Wk, H, nt, H, h0, h1, h2, h3, h4);
      int n  = nt * 16 + (lane & 15);
      int rb = mt0 * 16 + ((lane < 16) ? 0 : 8);
      v8f hr[5] = {h0, h1, h2, h3, h4};
#pragma unroll
      for (int kk = 0; kk < 5; ++kk)
#pragma unroll
        for (int v = 0; v < 8; ++v)
          hfb[(size_t)(rb + kk * 32 + v) * H + n] = hr[kk][v];
    }
    dir_barrier(bar, gen);

    // ---- P4a: scores s[b,l] = sum_h v*tanh(hf + q) (wave reductions) -------
    for (int idx = gw; idx < B * L; idx += NWAVE) {
      int b = idx / L;
      float sum = 0.f;
      for (int h = lane; h < H; h += 32)
        sum += vv[h] * tanhf(hfb[(size_t)idx * H + h] + qbuf[b * H + h]);
#pragma unroll
      for (int off = 16; off; off >>= 1) sum += __shfl_xor(sum, off, 32);
      if (lane == 0) sbuf[idx] = sum;
    }
    dir_barrier(bar, gen);

    // ---- P4b: softmax over L, d = sum_l a*astates, cP[:,H:] = d ------------
    for (int i = tidd; i < B * H; i += NT) {
      int b = i / H, h = i % H;
      float s0 = sbuf[b * L + 0], s1 = sbuf[b * L + 1], s2 = sbuf[b * L + 2],
            s3 = sbuf[b * L + 3], s4 = sbuf[b * L + 4];
      float mx = fmaxf(fmaxf(fmaxf(s0, s1), fmaxf(s2, s3)), s4);
      float e0 = __expf(s0 - mx), e1 = __expf(s1 - mx), e2 = __expf(s2 - mx),
            e3 = __expf(s3 - mx), e4 = __expf(s4 - mx);
      float inv = 1.0f / (e0 + e1 + e2 + e3 + e4);
      float dsum = (e0 * (float)ast[(size_t)(b * L + 0) * H + h] +
                    e1 * (float)ast[(size_t)(b * L + 1) * H + h] +
                    e2 * (float)ast[(size_t)(b * L + 2) * H + h] +
                    e3 * (float)ast[(size_t)(b * L + 3) * H + h] +
                    e4 * (float)ast[(size_t)(b * L + 4) * H + h]) * inv;
      dbuf[i] = dsum;
      cP[b * 2 * H + H + h] = (bf16)dsum;
    }
    dir_barrier(bar, gen);

    // ---- P5a: out = cP @ Wp + bp (128 tiles) -------------------------------
    {
      int mt = gw >> 6, nt = gw & 63;
      v8f acc = wmma_gemm(cP, 2 * H, mt * 16, Wp, H, nt, 2 * H);
      int n  = nt * 16 + (lane & 15);
      int rb = mt * 16 + ((lane < 16) ? 0 : 8);
      float bn = bp[n];
#pragma unroll
      for (int v = 0; v < 8; ++v)
        outb[(rb + v) * H + n] = acc[v] + bn;
    }
    dir_barrier(bar, gen);

    // ---- P5b: masked commit: S += out, attns=d, shift+append astates -------
    for (int i = tidd; i < B * H; i += NT) {
      int b = i / H, h = i % H;
      if (t < (length[b] / 4 + 1)) {
        float o = outb[i];
        Ssum[i] += o;                    // masked running sum (pooling identity)
        atbf[i] = (bf16)dbuf[i];
#pragma unroll
        for (int l = 0; l < L - 1; ++l)
          ast[(size_t)(b * L + l) * H + h] = ast[(size_t)(b * L + l + 1) * H + h];
        ast[(size_t)(b * L + L - 1) * H + h] = (bf16)o;
      }
    }
    dir_barrier(bar, gen);
  }
}

// --------------------------- final pooled logits ---------------------------
__global__ void logits_kernel(const char* ws, const float* __restrict__ Wlog,
                              const float* __restrict__ blog,
                              const int* __restrict__ length,
                              float* __restrict__ out) {
  int i = blockIdx.x * blockDim.x + threadIdx.x;
  if (i >= B * C) return;
  int b = i / C, c = i % C;
  const float* S0 = (const float*)(ws + OFF_S);
  const float* S1 = S0 + (size_t)B * H;
  float acc = 0.f;
  for (int h = 0; h < H; ++h)
    acc += S0[b * H + h] * Wlog[(size_t)h * C + c] +
           S1[b * H + h] * Wlog[(size_t)(H + h) * C + c];
  int sl = length[b] / 4 + 1;
  out[i] = acc / (float)T + (float)sl * blog[c] / (float)T;
}

// ------------------------------- launcher ----------------------------------
extern "C" void kernel_launch(void* const* d_in, const int* in_sizes, int n_in,
                              void* d_out, int out_size, void* d_ws, size_t ws_size,
                              hipStream_t stream) {
  (void)in_sizes; (void)n_in; (void)out_size; (void)ws_size;
  const float* x      = (const float*)d_in[0];
  const int*   length = (const int*)d_in[1];
  char* ws = (char*)d_ws;
  bf16* WB = (bf16*)(ws + OFF_W);

  zero_ws_kernel<<<256, 256, 0, stream>>>(ws);

  struct Job { const float* s; size_t off; int K, N; };
  const Job jobs[10] = {
      {(const float*)d_in[2],  0 * EW_DIR + EW_I, FH,    F},
      {(const float*)d_in[4],  0 * EW_DIR + EW_L, FH,    4 * H},
      {(const float*)d_in[6],  0 * EW_DIR + EW_K, H,     H},
      {(const float*)d_in[7],  0 * EW_DIR + EW_Q, 2 * H, H},
      {(const float*)d_in[10], 0 * EW_DIR + EW_P, 2 * H, H},
      {(const float*)d_in[12], 1 * EW_DIR + EW_I, FH,    F},
      {(const float*)d_in[14], 1 * EW_DIR + EW_L, FH,    4 * H},
      {(const float*)d_in[16], 1 * EW_DIR + EW_K, H,     H},
      {(const float*)d_in[17], 1 * EW_DIR + EW_Q, 2 * H, H},
      {(const float*)d_in[20], 1 * EW_DIR + EW_P, 2 * H, H},
  };
  for (int j = 0; j < 10; ++j) {
    size_t n = (size_t)jobs[j].K * jobs[j].N;
    int blocks = (int)((n + 255) / 256);
    if (blocks > 2048) blocks = 2048;
    pack_weight_kernel<<<blocks, 256, 0, stream>>>(jobs[j].s, WB + jobs[j].off,
                                                   jobs[j].K, jobs[j].N);
  }
  build_x_kernel<<<2048, 256, 0, stream>>>(x, length, (bf16*)(ws + OFF_XBF));

  recurrence_kernel<<<2 * NWG, THREADS, 0, stream>>>(
      ws, length,
      (const float*)d_in[3],  (const float*)d_in[5],  (const float*)d_in[8],
      (const float*)d_in[9],  (const float*)d_in[11],
      (const float*)d_in[13], (const float*)d_in[15], (const float*)d_in[18],
      (const float*)d_in[19], (const float*)d_in[21]);

  logits_kernel<<<(B * C + 255) / 256, 256, 0, stream>>>(
      ws, (const float*)d_in[22], (const float*)d_in[23], length,
      (float*)d_out);
}
